// STFNet_84198538871411
// MI455X (gfx1250) — matrix operand order repaired and checked by reading, not compiled, not run on hardware
//
#include <hip/hip_runtime.h>

typedef __attribute__((ext_vector_type(16))) _Float16 v16h;
typedef __attribute__((ext_vector_type(8)))  float    v8f;
typedef __attribute__((ext_vector_type(4)))  unsigned int v4u;
typedef __attribute__((ext_vector_type(8)))  int      v8i;
typedef __attribute__((ext_vector_type(4)))  int      v4i;

#define LDH  72    // padded LDS stride in halves (36*r mod 64 distinct -> conflict-free)
#define AUGW 132   // padded stride for 64x128 augmented Gauss-Jordan array

#if defined(__has_builtin)
#if __has_builtin(__builtin_amdgcn_tensor_load_to_lds) && __has_builtin(__builtin_amdgcn_s_wait_tensorcnt)
#define USE_TDM 1
#endif
#endif
#ifndef USE_TDM
#define USE_TDM 0
#endif

__device__ __forceinline__ v8f vzero8() { v8f z = {}; return z; }

__device__ __forceinline__ v8f wmma_f16(v16h a, v16h b, v8f c) {
  // (neg_a, A, neg_b, B, c_mod, C, reuse_a, reuse_b) -> v_wmma_f32_16x16x32_f16
  return __builtin_amdgcn_wmma_f32_16x16x32_f16(false, a, false, b, (short)0, c, false, false);
}

// ---------------------------------------------------------------------------
// GEMM: C = relu(A(MxK,f32) @ B(KxN,f32) + bias), output f32 or f16.
// 512 threads = 16 wave32; block tile 128x128; each wave owns a 2x2 grid of
// 16x16 accumulators (4 WMMA per fragment pair -> 2 ds_load_b128 per WMMA).
// B is staged TRANSPOSED in LDS so B-fragments are contiguous per lane.
// ---------------------------------------------------------------------------
template <bool HALF_OUT>
__global__ __launch_bounds__(512)
void gemm_bias_relu(const float* __restrict__ A, const float* __restrict__ B,
                    const float* __restrict__ bias,
                    float* __restrict__ Cf, _Float16* __restrict__ Ch,
                    int M, int N, int K)
{
  __shared__ _Float16 sA [128 * LDH];   // 128 rows x 64 k
  __shared__ _Float16 sBt[128 * LDH];   // 128 cols x 64 k (transposed)

  const int tid  = threadIdx.x;
  const int lane = tid & 31;
  const int wave = tid >> 5;
  const int wi = wave >> 2, wj = wave & 3;     // 4x4 waves over 128x128
  const int hi = lane >> 4, lm = lane & 15;
  const int m0 = blockIdx.y << 7, n0 = blockIdx.x << 7;

  v8f acc00 = vzero8(), acc01 = vzero8(), acc10 = vzero8(), acc11 = vzero8();

  for (int kb = 0; kb < K; kb += 64) {
#pragma unroll
    for (int i = 0; i < 16; ++i) {             // 8192 elems each, coalesced reads
      int idx = tid + (i << 9);
      int rA = idx >> 6, cA = idx & 63;        // A: 128 x 64
      sA[rA * LDH + cA] = (_Float16)A[(size_t)(m0 + rA) * K + kb + cA];
      int rB = idx >> 7, cB = idx & 127;       // B: 64 x 128 -> store transposed
      sBt[cB * LDH + rB] = (_Float16)B[(size_t)(kb + rB) * N + n0 + cB];
    }
    __syncthreads();
#pragma unroll
    for (int kk = 0; kk < 2; ++kk) {
      v16h a0, a1, b0, b1;
#pragma unroll
      for (int h = 0; h < 16; ++h) {
        int j  = h >> 1;
        // A 16x32 f16 layout: K = 16*(lane/16) + h
        int ka = (kk << 5) + (hi << 4) + h;
        a0[h] = sA[(wi * 32      + lm) * LDH + ka];
        a1[h] = sA[(wi * 32 + 16 + lm) * LDH + ka];
        // B 32x16 f16 layout: K = 16*(lane/16) + h, N = lane%16 (from sBt rows)
        int kb2 = (kk << 5) + (hi << 4) + h;
        b0[h] = sBt[(wj * 32      + lm) * LDH + kb2];
        b1[h] = sBt[(wj * 32 + 16 + lm) * LDH + kb2];
      }
      acc00 = wmma_f16(a0, b0, acc00);
      acc01 = wmma_f16(a0, b1, acc01);
      acc10 = wmma_f16(a1, b0, acc10);
      acc11 = wmma_f16(a1, b1, acc11);
    }
    __syncthreads();
  }

#pragma unroll
  for (int ta = 0; ta < 2; ++ta) {
#pragma unroll
    for (int tb = 0; tb < 2; ++tb) {
      v8f acc = (ta == 0) ? (tb == 0 ? acc00 : acc01) : (tb == 0 ? acc10 : acc11);
#pragma unroll
      for (int j = 0; j < 8; ++j) {            // C/D: M = j + 8*(lane/16), N = lane%16
        int m = m0 + wi * 32 + ta * 16 + j + (hi << 3);
        int n = n0 + wj * 32 + tb * 16 + lm;
        float v = acc[j] + bias[n];
        v = v > 0.f ? v : 0.f;
        if (HALF_OUT) Ch[(size_t)m * N + n] = (_Float16)v;
        else          Cf[(size_t)m * N + n] = v;
      }
    }
  }
}

// ---------------------------------------------------------------------------
// Sequential Kalman scan: single persistent workgroup (512 thr / 16 waves).
// Per step: TDM-loads T_t into LDS (padded via D# pad fields), 4 WMMA-backed
// 64x64x64 matmuls, LDS Gauss-Jordan inversion, matvec updates.
// LDS buffers are time-multiplexed:
//   sB0: T_t        -> K
//   sB1: P (f16)    -> P_pred (f16) -> P_upd (f16, next step's P)
//   sB2: M1 = T@P   -> S^-1
// ---------------------------------------------------------------------------
__global__ __launch_bounds__(512)
void kalman_scan(const float* __restrict__ Y, const _Float16* __restrict__ T,
                 const float* __restrict__ kvars,
                 float* __restrict__ alpha_out, float* __restrict__ P_out)
{
  __shared__ _Float16 sB0[64 * LDH];
  __shared__ _Float16 sB1[64 * LDH];
  __shared__ _Float16 sB2[64 * LDH];
  __shared__ float sAug[64][AUGW];             // [S | I] -> [I | S^-1]
  __shared__ float sFac[64], sAlphaPred[64], sAlpha[64], sVec[64], sY[64];

  const int tid  = threadIdx.x;
  const int lane = tid & 31;
  const int wave = tid >> 5;
  const int ti = wave >> 2, tj = wave & 3;
  const int hi = lane >> 4, lm = lane & 15;
  const float q  = kvars[0];                   // predict noise
  const float hN = kvars[1];                   // measurement noise

  // ---- init: alpha0 = 1, P0 = I ----
  if (tid < 64) { sAlpha[tid] = 1.0f; alpha_out[tid] = 1.0f; }
  for (int i = tid; i < 4096; i += 512) {
    int r = i >> 6, c = i & 63;
    float v = (r == c) ? 1.0f : 0.0f;
    P_out[i] = v;
    sB1[r * LDH + c] = (_Float16)v;
  }
  __syncthreads();

  for (int t = 1; t < 4096; ++t) {
    const _Float16* Tg = T + (size_t)(t - 1) * 4096;
#if USE_TDM
    // Tensor Data Mover: 2D tile 64x64 of 2-byte elems, global -> LDS, with
    // hardware padding of 4 DWORDs every 32 DWORDs to reproduce the LDH=72
    // padded row stride. Issued once (wave 0), fenced by TENSORcnt + barrier.
    if (wave == 0) {
      unsigned lds_base = (unsigned)(unsigned long long)(uintptr_t)&sB0[0];
      unsigned long long ga = (unsigned long long)(uintptr_t)Tg;
      v4u g0;
      g0[0] = 1u;                                            // count=1 valid descriptor
      g0[1] = lds_base;                                      // lds_addr
      g0[2] = (unsigned)(ga & 0xFFFFFFFFu);                  // global_addr[31:0]
      g0[3] = (unsigned)((ga >> 32) & 0x1FFFFFFu) | (2u << 30); // addr[56:32] | type=2
      v8i g1;
      g1[0] = (int)((1u << 16)        // data_size = 2 bytes
                  | (1u << 20)        // pad_enable
                  | (4u << 22)        // pad_interval: 2^(4+1)=32 DWORDs (one 128B row)
                  | (3u << 25));      // pad_amount: 4 DWORDs (16B = 8 halves)
      g1[1] = (int)(64u << 16);                              // tensor_dim0 = 64 (lo16)
      g1[2] = (int)(4096u << 16);                            // dim0 hi=0 | tensor_dim1 lo16 = 4096
      g1[3] = (int)(64u << 16);                              // dim1 hi=0 | tile_dim0 = 64
      g1[4] = (int)64;                                       // tile_dim1 = 64, tile_dim2 = 0
      g1[5] = (int)64;                                       // tensor_dim0_stride = 64 (lo32)
      g1[6] = 0;                                             // stride0 hi | stride1 lo
      g1[7] = 0;
      v4i gz = {0, 0, 0, 0};
#if defined(__clang_major__) && (__clang_major__ >= 23)
      v8i gz8 = {0, 0, 0, 0, 0, 0, 0, 0};
      __builtin_amdgcn_tensor_load_to_lds(g0, g1, gz, gz, gz8, 0);
#else
      __builtin_amdgcn_tensor_load_to_lds(g0, g1, gz, gz, 0);
#endif
      __builtin_amdgcn_s_wait_tensorcnt(0);
    }
#else
    for (int i = tid; i < 4096; i += 512)
      sB0[(i >> 6) * LDH + (i & 63)] = Tg[i];
#endif
    if (tid < 64) sY[tid] = Y[(size_t)t * 64 + tid];
    if (t < 4095) __builtin_prefetch((const void*)(Tg + 4096 + tid * 8), 0, 3);
    __syncthreads();

    // ---- alpha_pred = T @ alpha (64 threads) ----
    if (tid < 64) {
      float s = 0.f;
#pragma unroll 8
      for (int k = 0; k < 64; ++k) s += (float)sB0[tid * LDH + k] * sAlpha[k];
      sAlphaPred[tid] = s;
    }

    // ---- M1 = T @ P -> sB2 ----
    v8f acc = vzero8();
#pragma unroll
    for (int kk = 0; kk < 2; ++kk) {
      v16h a, b;
#pragma unroll
      for (int h = 0; h < 16; ++h) {
        int j  = h >> 1;
        int ka = (kk << 5) + (hi << 4) + h;
        a[h] = sB0[(ti * 16 + lm) * LDH + ka];
        b[h] = sB1[((kk << 5) + (hi << 4) + h) * LDH + tj * 16 + lm];
      }
      acc = wmma_f16(a, b, acc);
    }
#pragma unroll
    for (int j = 0; j < 8; ++j)
      sB2[(ti * 16 + j + (hi << 3)) * LDH + tj * 16 + lm] = (_Float16)acc[j];
    __syncthreads();

    // ---- P_pred = M1 @ T^T + q*I -> sB1(f16) and sAug[:, :64] (S = P_pred + h*I) ----
    acc = vzero8();
#pragma unroll
    for (int kk = 0; kk < 2; ++kk) {
      v16h a, b;
#pragma unroll
      for (int h = 0; h < 16; ++h) {
        int j  = h >> 1;
        int ka = (kk << 5) + (hi << 4) + h;
        a[h] = sB2[(ti * 16 + lm) * LDH + ka];
        b[h] = sB0[(tj * 16 + lm) * LDH + (kk << 5) + (hi << 4) + h];   // T^T rows
      }
      acc = wmma_f16(a, b, acc);
    }
#pragma unroll
    for (int j = 0; j < 8; ++j) {
      int m = ti * 16 + j + (hi << 3);
      int n = tj * 16 + lm;
      float v = acc[j] + (m == n ? q : 0.f);
      sB1[m * LDH + n] = (_Float16)v;                 // P_pred (f16 operand)
      sAug[m][n]       = v + (m == n ? hN : 0.f);     // S
      sAug[m][64 + n]  = (m == n) ? 1.0f : 0.0f;      // I
    }
    __syncthreads();

    // ---- Gauss-Jordan: [S | I] -> [I | S^-1] ----
    for (int p = 0; p < 64; ++p) {
      if (tid < 64) sFac[tid] = sAug[tid][p];
      __syncthreads();
      float pinv = 1.0f / sFac[p];
      if (tid < 128) sAug[p][tid] *= pinv;
      __syncthreads();
      {
        int r = tid >> 3, c0 = (tid & 7) << 4;        // 8 thr/row, 16 cols each
        if (r != p) {
          float f = sFac[r];
#pragma unroll
          for (int c = 0; c < 16; ++c)
            sAug[r][c0 + c] -= f * sAug[p][c0 + c];
        }
      }
      __syncthreads();
    }
    // S^-1 -> sB2 (f16)
    for (int i = tid; i < 4096; i += 512)
      sB2[(i >> 6) * LDH + (i & 63)] = (_Float16)sAug[i >> 6][64 + (i & 63)];
    __syncthreads();

    // ---- K = P_pred @ S^-1 -> sB0 (overwrites T, now dead) ----
    acc = vzero8();
#pragma unroll
    for (int kk = 0; kk < 2; ++kk) {
      v16h a, b;
#pragma unroll
      for (int h = 0; h < 16; ++h) {
        int j  = h >> 1;
        int ka = (kk << 5) + (hi << 4) + h;
        a[h] = sB1[(ti * 16 + lm) * LDH + ka];
        b[h] = sB2[((kk << 5) + (hi << 4) + h) * LDH + tj * 16 + lm];
      }
      acc = wmma_f16(a, b, acc);
    }
#pragma unroll
    for (int j = 0; j < 8; ++j)
      sB0[(ti * 16 + j + (hi << 3)) * LDH + tj * 16 + lm] = (_Float16)acc[j];
    __syncthreads();

    // ---- KP = K @ P_pred ; P_upd = I - KP -> global + sB1 (next step's P) ----
    acc = vzero8();
#pragma unroll
    for (int kk = 0; kk < 2; ++kk) {
      v16h a, b;
#pragma unroll
      for (int h = 0; h < 16; ++h) {
        int j  = h >> 1;
        int ka = (kk << 5) + (hi << 4) + h;
        a[h] = sB0[(ti * 16 + lm) * LDH + ka];
        b[h] = sB1[((kk << 5) + (hi << 4) + h) * LDH + tj * 16 + lm];
      }
      acc = wmma_f16(a, b, acc);
    }
    if (tid < 64) sVec[tid] = sY[tid] - sAlphaPred[tid];
    __syncthreads();                                  // all sB1 reads done before overwrite
#pragma unroll
    for (int j = 0; j < 8; ++j) {
      int m = ti * 16 + j + (hi << 3);
      int n = tj * 16 + lm;
      float v = ((m == n) ? 1.0f : 0.0f) - acc[j];
      sB1[m * LDH + n] = (_Float16)v;
      P_out[(size_t)t * 4096 + m * 64 + n] = v;
    }

    // ---- alpha_upd = alpha_pred + K @ (y - alpha_pred) ----
    if (tid < 64) {
      float s = sAlphaPred[tid];
#pragma unroll 8
      for (int k = 0; k < 64; ++k) s += (float)sB0[tid * LDH + k] * sVec[k];
      sAlpha[tid] = s;
      alpha_out[(size_t)t * 64 + tid] = s;
    }
    __syncthreads();                                  // sAlpha/sB1 ready; sB0 free
  }
}

// ---------------------------------------------------------------------------
// Launch: 3 WMMA GEMMs (MLP) then the persistent sequential Kalman scan.
// Workspace: H1(8MB f32) | H2(8MB f32) | T(32MB f16) = 48MB.
// d_out: alpha_est (4096*64 f32) then P_est (4096*64*64 f32).
// ---------------------------------------------------------------------------
extern "C" void kernel_launch(void* const* d_in, const int* in_sizes, int n_in,
                              void* d_out, int out_size, void* d_ws, size_t ws_size,
                              hipStream_t stream) {
  (void)in_sizes; (void)n_in; (void)out_size; (void)ws_size;

  const float* Y  = (const float*)d_in[0];
  const float* Z  = (const float*)d_in[1];
  const float* W1 = (const float*)d_in[2];
  const float* b1 = (const float*)d_in[3];
  const float* W2 = (const float*)d_in[4];
  const float* b2 = (const float*)d_in[5];
  const float* W3 = (const float*)d_in[6];
  const float* b3 = (const float*)d_in[7];
  const float* kv = (const float*)d_in[8];

  float* out       = (float*)d_out;
  float* alpha_out = out;
  float* P_out     = out + (size_t)4096 * 64;

  float*    H1 = (float*)d_ws;
  float*    H2 = H1 + (size_t)4096 * 512;
  _Float16* T  = (_Float16*)(H2 + (size_t)4096 * 512);

  dim3 blk(512, 1, 1);
  // grid = (N/128, M/128)
  gemm_bias_relu<false><<<dim3(4, 32),  blk, 0, stream>>>(Z,  W1, b1, H1, (_Float16*)nullptr, 4096, 512,  128);
  gemm_bias_relu<false><<<dim3(4, 32),  blk, 0, stream>>>(H1, W2, b2, H2, (_Float16*)nullptr, 4096, 512,  512);
  gemm_bias_relu<true ><<<dim3(32, 32), blk, 0, stream>>>(H2, W3, b3, (float*)nullptr, T,     4096, 4096, 512);
  kalman_scan<<<dim3(1), blk, 0, stream>>>(Y, T, kv, alpha_out, P_out);
}
